// ParallelWaveGANGenerator_77610059038928
// MI455X (gfx1250) — matrix operand order repaired and checked by reading, not compiled, not run on hardware
//
#include <hip/hip_runtime.h>
#include <hip/hip_bf16.h>
#include <math.h>

// ---------------------------------------------------------------------------
// Types for CDNA5 WMMA (wave32): D(16x16 f32) = A(16x32 bf16) * B(32x16 bf16) + C
// ---------------------------------------------------------------------------
typedef __attribute__((ext_vector_type(16))) __bf16      v16bf;
typedef __attribute__((ext_vector_type(8)))  float       v8f;
typedef __attribute__((ext_vector_type(4)))  unsigned int u32x4;

union Frag16 { v16bf v; u32x4 q[2]; };

__device__ inline unsigned short f2bf(float f) {
    union { float f; unsigned u; } v; v.f = f;
    unsigned r = v.u + 0x7FFFu + ((v.u >> 16) & 1u);
    return (unsigned short)(r >> 16);
}
__device__ inline float bf2f(unsigned short h) {
    union { unsigned u; float f; } v; v.u = ((unsigned)h) << 16;
    return v.f;
}

// B-matrix fragment (32x16 bf16): element e -> K = kbase+e (contiguous 16 halfs).
__device__ inline v16bf load_frag16(const unsigned short* p) {
    Frag16 f;
    f.q[0] = *(const u32x4*)(p);
    f.q[1] = *(const u32x4*)(p + 8);
    return f.v;
}
// A-matrix fragment (16x32 bf16) from row-major W[row][K]:
//  lane&15 = M row; khalf = (lane>=16)*16; elem e -> K=k0+khalf+e
__device__ inline v16bf load_fragA(const unsigned short* __restrict__ W,
                                   int row0, int K, int k0, int lane) {
    int m = lane & 15;
    int khalf = (lane >> 4) << 4;
    const unsigned short* p = W + (size_t)(row0 + m) * K + k0 + khalf;
    Frag16 f;
    f.q[0] = *(const u32x4*)(p);
    f.q[1] = *(const u32x4*)(p + 8);
    return f.v;
}

#define WMMA_BF16(A, B, C) \
    __builtin_amdgcn_wmma_f32_16x16x32_bf16(false, (A), false, (B), (short)0, (C), false, false)

// ---------------------------------------------------------------------------
// Constants
// ---------------------------------------------------------------------------
#define BATCH   2
#define TLEN    32768
#define TFRAMES 128
#define RC      64
#define GC      128
#define SC      64
#define ACPAD   96      // 80 aux channels padded to 96 for K-tiling
#define NLAYERS 30
#define TT      64      // timestep tile per block (4 N-tiles -> 4x A-fragment reuse)
#define NTILES  (TT / 16)
#define SQRT_HALF 0.70710678118654752f
#define SKIP_SCALE 0.18257418583505536f  // sqrt(1/30)

// ---------------------------------------------------------------------------
// c0 = W_cin @ c     (B,80,128)
// ---------------------------------------------------------------------------
__global__ void k_cin(const float* __restrict__ c, const float* __restrict__ Wcin,
                      float* __restrict__ out, int total) {
    int idx = blockIdx.x * blockDim.x + threadIdx.x;
    if (idx >= total) return;
    int f = idx % TFRAMES;
    int r = idx / TFRAMES;
    int o = r % 80;
    int b = r / 80;
    float acc = 0.f;
    const float* cin = c + ((size_t)b * 80) * TFRAMES + f;
    const float* w = Wcin + (size_t)o * 80;
    for (int ci = 0; ci < 80; ++ci) acc += w[ci] * cin[(size_t)ci * TFRAMES];
    out[idx] = acc;
}

// ---------------------------------------------------------------------------
// One upsample stage: repeat x4 then 9-tap conv, float->float, layout [b][ch][W]
// ---------------------------------------------------------------------------
__global__ void k_upstage(const float* __restrict__ in, float* __restrict__ out,
                          const float* __restrict__ w9, int Win, int total) {
    int idx = blockIdx.x * blockDim.x + threadIdx.x;
    if (idx >= total) return;
    int Wout = Win * 4;
    int t  = idx % Wout;
    int bc = idx / Wout;
    const float* src = in + (size_t)bc * Win;
    float acc = 0.f;
    #pragma unroll
    for (int k = 0; k < 9; ++k) {
        int j = t - 4 + k;
        if (j >= 0 && j < Wout) acc += w9[k] * src[j >> 2];
    }
    out[idx] = acc;
}

// Last stage: writes bf16 c_up, layout [b][t][96] (ch >= 80 zero padded)
__global__ void k_upstage_last(const float* __restrict__ in, unsigned short* __restrict__ cup,
                               const float* __restrict__ w9, int Win, int total) {
    int idx = blockIdx.x * blockDim.x + threadIdx.x;
    if (idx >= total) return;
    int t = idx % TLEN;
    int r = idx / TLEN;
    int ch = r % ACPAD;
    int b  = r / ACPAD;
    float acc = 0.f;
    if (ch < 80) {
        int Wout = Win * 4;
        const float* src = in + ((size_t)b * 80 + ch) * Win;
        #pragma unroll
        for (int k = 0; k < 9; ++k) {
            int j = t - 4 + k;
            if (j >= 0 && j < Wout) acc += w9[k] * src[j >> 2];
        }
    }
    cup[((size_t)b * TLEN + t) * ACPAD + ch] = f2bf(acc);
}

// ---------------------------------------------------------------------------
// Weight repack to bf16, K-major rows for WMMA A-fragments
// ---------------------------------------------------------------------------
__global__ void k_prep_wd(const float* __restrict__ Wd, unsigned short* __restrict__ dst, int total) {
    int idx = blockIdx.x * blockDim.x + threadIdx.x;
    if (idx >= total) return;
    int k = idx % 192;                 // K = tap*64 + cin
    int g = (idx / 192) % GC;
    int i = idx / (192 * GC);
    int tap = k >> 6, cin = k & 63;
    dst[idx] = f2bf(Wd[(((size_t)i * GC + g) * RC + cin) * 3 + tap]);
}
__global__ void k_prep_wa(const float* __restrict__ Wa, unsigned short* __restrict__ dst, int total) {
    int idx = blockIdx.x * blockDim.x + threadIdx.x;
    if (idx >= total) return;
    int k = idx % ACPAD;
    int g = (idx / ACPAD) % GC;
    int i = idx / (ACPAD * GC);
    dst[idx] = (k < 80) ? f2bf(Wa[((size_t)i * GC + g) * 80 + k]) : (unsigned short)0;
}
__global__ void k_prep_cvt(const float* __restrict__ src, unsigned short* __restrict__ dst, int total) {
    int idx = blockIdx.x * blockDim.x + threadIdx.x;
    if (idx >= total) return;
    dst[idx] = f2bf(src[idx]);
}

// ---------------------------------------------------------------------------
// First 1x1 (1->64) + zero skips.  h layout [b][t][64] bf16, skips [b][t][64] f32
// ---------------------------------------------------------------------------
__global__ void k_first(const float* __restrict__ x, const float* __restrict__ Wf,
                        const float* __restrict__ bf, unsigned short* __restrict__ h,
                        float* __restrict__ skips, int total) {
    int idx = blockIdx.x * blockDim.x + threadIdx.x;   // idx = b*T + t
    if (idx >= total) return;
    float xv = x[idx];
    size_t base = (size_t)idx * RC;
    for (int ch = 0; ch < RC; ++ch) {
        h[base + ch] = f2bf(Wf[ch] * xv + bf[ch]);
        skips[base + ch] = 0.f;
    }
}

// ---------------------------------------------------------------------------
// Fused WaveNet layer (the WMMA workhorse).
//   g = Wd *_dilated h + bd + Wa * c_up          (128ch x 64t tile)
//   z = tanh(g[0:64]) * sigmoid(g[64:128])
//   skips += Ws*z + bs ;  h_out = (Wr*z + br + h) * sqrt(0.5)
// Block: 128 threads (4 waves), one (batch, 64-timestep) tile.
// Each wave: gate tiles {wv, wv+4} x 4 N-tiles; skip/res tile wv x 4 N-tiles.
// A fragments reused 4x across N-tiles (cuts weight L2 traffic 4x).
// ---------------------------------------------------------------------------
__global__ __launch_bounds__(128)
void k_layer(const unsigned short* __restrict__ h_in,
             unsigned short* __restrict__ h_out,
             float* __restrict__ skips,
             const unsigned short* __restrict__ cup,
             const unsigned short* __restrict__ wd,   // [128][192]
             const unsigned short* __restrict__ wa,   // [128][96]
             const unsigned short* __restrict__ wsk,  // [64][64]
             const unsigned short* __restrict__ wr,   // [64][64]
             const float* __restrict__ bd,            // [128]
             const float* __restrict__ bs,            // [64]
             const float* __restrict__ br,            // [64]
             int dilation, int T) {
    __shared__ __align__(16) unsigned short lds_h[3][TT][RC];   // 24KB: 3 taps, time-major
    __shared__ __align__(16) unsigned short lds_c[TT][ACPAD];   // 12KB
    __shared__ __align__(16) unsigned short lds_z[TT][RC];      //  8KB

    const int tid  = threadIdx.x;
    const int lane = tid & 31;
    const int wv   = tid >> 5;                 // wave 0..3
    const int tiles_per_batch = T / TT;
    const int b  = blockIdx.x / tiles_per_batch;
    const int t0 = (blockIdx.x % tiles_per_batch) * TT;

    __builtin_prefetch(wd, 0, 1);              // global_prefetch_b8: pull weights toward WGP

    // ---- Stage h tiles (3 taps) into LDS, 128B rows via B128 ----
    const unsigned short* hb = h_in + (size_t)b * T * RC;
    for (int idx = tid; idx < 3 * TT * 8; idx += 128) {
        int seg = idx & 7;
        int row = idx >> 3;                    // 0..191
        int tap = row >> 6;
        int n   = row & (TT - 1);
        int t   = t0 + n + (tap - 1) * dilation;
        u32x4 val = {0u, 0u, 0u, 0u};
        if (t >= 0 && t < T) val = *(const u32x4*)(hb + (size_t)t * RC + seg * 8);
        *(u32x4*)(&lds_h[tap][n][seg * 8]) = val;
    }
    // ---- Stage c_up tile ----
    const unsigned short* cb = cup + (size_t)b * T * ACPAD;
    for (int idx = tid; idx < TT * 12; idx += 128) {
        int seg = idx % 12;
        int n   = idx / 12;
        *(u32x4*)(&lds_c[n][seg * 8]) = *(const u32x4*)(cb + (size_t)(t0 + n) * ACPAD + seg * 8);
    }
    __syncthreads();

    const int n16   = lane & 15;               // N within a 16-tile
    const int hi    = lane >> 4;               // 0 lo-half, 1 hi-half
    const int kbase = hi << 4;                 // B-frag K base offset

    // ---- Gate GEMM: 2 channel tiles x 4 N-tiles per wave ----
    v8f acc[2][NTILES];
    #pragma unroll
    for (int p = 0; p < 2; ++p) {
        int row0 = (wv + 4 * p) * 16;
        #pragma unroll
        for (int nt = 0; nt < NTILES; ++nt)
            #pragma unroll
            for (int r = 0; r < 8; ++r) acc[p][nt][r] = bd[row0 + r + 8 * hi];
    }
    // 9 k-steps: 0..5 dilated conv (K=192), 6..8 aux (K=96, zero-padded 80..95)
    #pragma unroll
    for (int ks = 0; ks < 9; ++ks) {
        v16bf bfr[NTILES];
        if (ks < 6) {
            int k0  = ks * 32;
            int tap = k0 >> 6;
            int kc  = (k0 & 63) + kbase;
            #pragma unroll
            for (int nt = 0; nt < NTILES; ++nt)
                bfr[nt] = load_frag16(&lds_h[tap][nt * 16 + n16][kc]);
        } else {
            int k0 = (ks - 6) * 32;
            #pragma unroll
            for (int nt = 0; nt < NTILES; ++nt)
                bfr[nt] = load_frag16(&lds_c[nt * 16 + n16][k0 + kbase]);
        }
        #pragma unroll
        for (int p = 0; p < 2; ++p) {
            int row0 = (wv + 4 * p) * 16;
            v16bf a = (ks < 6) ? load_fragA(wd, row0, 192, ks * 32, lane)
                               : load_fragA(wa, row0, ACPAD, (ks - 6) * 32, lane);
            #pragma unroll
            for (int nt = 0; nt < NTILES; ++nt)
                acc[p][nt] = WMMA_BF16(a, bfr[nt], acc[p][nt]);
        }
    }

    // ---- Gating: z = tanh(xa) * sigmoid(xb), z channels 16*wv .. 16*wv+15 ----
    #pragma unroll
    for (int nt = 0; nt < NTILES; ++nt) {
        #pragma unroll
        for (int r = 0; r < 8; ++r) {
            float za = tanhf(acc[0][nt][r]);
            float zb = 1.f / (1.f + expf(-acc[1][nt][r]));
            int ch = 16 * wv + r + 8 * hi;
            lds_z[nt * 16 + n16][ch] = f2bf(za * zb);
        }
    }
    __syncthreads();

    // ---- Skip & residual GEMMs: out tile wv, K=64 (2 k-steps), 4 N-tiles ----
    {
        int row0 = wv * 16;
        v8f sacc[NTILES], racc[NTILES];
        #pragma unroll
        for (int nt = 0; nt < NTILES; ++nt)
            #pragma unroll
            for (int r = 0; r < 8; ++r) {
                sacc[nt][r] = bs[row0 + r + 8 * hi];
                racc[nt][r] = br[row0 + r + 8 * hi];
            }
        #pragma unroll
        for (int ks = 0; ks < 2; ++ks) {
            int k0 = ks * 32;
            v16bf bz[NTILES];
            #pragma unroll
            for (int nt = 0; nt < NTILES; ++nt)
                bz[nt] = load_frag16(&lds_z[nt * 16 + n16][k0 + kbase]);
            v16bf as = load_fragA(wsk, row0, RC, k0, lane);
            #pragma unroll
            for (int nt = 0; nt < NTILES; ++nt)
                sacc[nt] = WMMA_BF16(as, bz[nt], sacc[nt]);
            v16bf ar = load_fragA(wr, row0, RC, k0, lane);
            #pragma unroll
            for (int nt = 0; nt < NTILES; ++nt)
                racc[nt] = WMMA_BF16(ar, bz[nt], racc[nt]);
        }
        #pragma unroll
        for (int nt = 0; nt < NTILES; ++nt) {
            #pragma unroll
            for (int r = 0; r < 8; ++r) {
                int ch = row0 + r + 8 * hi;
                int t  = t0 + nt * 16 + n16;
                size_t gidx = ((size_t)b * T + t) * RC + ch;
                skips[gidx] += sacc[nt][r];
                float hprev = bf2f(lds_h[1][nt * 16 + n16][ch]);   // center tap
                h_out[gidx] = f2bf((racc[nt][r] + hprev) * SQRT_HALF);
            }
        }
    }
}

// ---------------------------------------------------------------------------
// Head: y = Wl2 @ relu(Wl1 @ relu(skips * sqrt(1/30)) + bl1) + bl2
// ---------------------------------------------------------------------------
__global__ __launch_bounds__(256)
void k_final(const float* __restrict__ skips, const float* __restrict__ Wl1,
             const float* __restrict__ bl1, const float* __restrict__ Wl2,
             const float* __restrict__ bl2, float* __restrict__ out, int total) {
    __shared__ float sw[SC * SC];
    for (int i = threadIdx.x; i < SC * SC; i += 256) sw[i] = Wl1[i];
    __syncthreads();
    int idx = blockIdx.x * blockDim.x + threadIdx.x;   // b*T + t
    if (idx >= total) return;
    float s[SC];
    const float* sp = skips + (size_t)idx * SC;
    #pragma unroll 8
    for (int c = 0; c < SC; ++c) {
        float v = sp[c] * SKIP_SCALE;
        s[c] = v > 0.f ? v : 0.f;
    }
    float y = 0.f;
    for (int o = 0; o < SC; ++o) {
        float a = bl1[o];
        const float* wrow = &sw[o * SC];
        #pragma unroll 8
        for (int c = 0; c < SC; ++c) a += wrow[c] * s[c];
        y += Wl2[o] * (a > 0.f ? a : 0.f);
    }
    out[idx] = y + bl2[0];
}

// ---------------------------------------------------------------------------
// Host launcher
// ---------------------------------------------------------------------------
extern "C" void kernel_launch(void* const* d_in, const int* in_sizes, int n_in,
                              void* d_out, int out_size, void* d_ws, size_t ws_size,
                              hipStream_t stream) {
    (void)in_sizes; (void)n_in; (void)out_size; (void)ws_size;
    const float* x       = (const float*)d_in[0];
    const float* c       = (const float*)d_in[1];
    const float* W_first = (const float*)d_in[2];
    const float* b_first = (const float*)d_in[3];
    const float* W_cin   = (const float*)d_in[4];
    const float* W_up    = (const float*)d_in[5];
    const float* Wd      = (const float*)d_in[6];
    const float* bd      = (const float*)d_in[7];
    const float* Wa      = (const float*)d_in[8];
    const float* Ws      = (const float*)d_in[9];
    const float* bs      = (const float*)d_in[10];
    const float* Wr      = (const float*)d_in[11];
    const float* br      = (const float*)d_in[12];
    const float* Wl1     = (const float*)d_in[13];
    const float* bl1     = (const float*)d_in[14];
    const float* Wl2     = (const float*)d_in[15];
    const float* bl2     = (const float*)d_in[16];
    float* out = (float*)d_out;

    const int B = BATCH, T = TLEN;
    char* ws = (char*)d_ws;
    size_t off = 0;
    auto alloc = [&](size_t bytes) -> char* {
        char* p = ws + off;
        off = (off + bytes + 255) & ~(size_t)255;
        return p;
    };
    float* c0 = (float*)alloc((size_t)B * 80 * 128 * 4);
    float* u1 = (float*)alloc((size_t)B * 80 * 512 * 4);
    float* u2 = (float*)alloc((size_t)B * 80 * 2048 * 4);
    float* u3 = (float*)alloc((size_t)B * 80 * 8192 * 4);
    unsigned short* cup = (unsigned short*)alloc((size_t)B * T * ACPAD * 2);
    unsigned short* h0  = (unsigned short*)alloc((size_t)B * T * RC * 2);
    unsigned short* h1  = (unsigned short*)alloc((size_t)B * T * RC * 2);
    float* skips        = (float*)alloc((size_t)B * T * RC * 4);
    unsigned short* wdq = (unsigned short*)alloc((size_t)NLAYERS * GC * 192 * 2);
    unsigned short* waq = (unsigned short*)alloc((size_t)NLAYERS * GC * ACPAD * 2);
    unsigned short* wsq = (unsigned short*)alloc((size_t)NLAYERS * SC * RC * 2);
    unsigned short* wrq = (unsigned short*)alloc((size_t)NLAYERS * SC * RC * 2);

    int n;
    n = B * 80 * 128;   k_cin<<<(n + 255) / 256, 256, 0, stream>>>(c, W_cin, c0, n);
    n = B * 80 * 512;   k_upstage<<<(n + 255) / 256, 256, 0, stream>>>(c0, u1, W_up + 0,  128,  n);
    n = B * 80 * 2048;  k_upstage<<<(n + 255) / 256, 256, 0, stream>>>(u1, u2, W_up + 9,  512,  n);
    n = B * 80 * 8192;  k_upstage<<<(n + 255) / 256, 256, 0, stream>>>(u2, u3, W_up + 18, 2048, n);
    n = B * ACPAD * T;  k_upstage_last<<<(n + 255) / 256, 256, 0, stream>>>(u3, cup, W_up + 27, 8192, n);

    n = NLAYERS * GC * 192;   k_prep_wd<<<(n + 255) / 256, 256, 0, stream>>>(Wd, wdq, n);
    n = NLAYERS * GC * ACPAD; k_prep_wa<<<(n + 255) / 256, 256, 0, stream>>>(Wa, waq, n);
    n = NLAYERS * SC * RC;    k_prep_cvt<<<(n + 255) / 256, 256, 0, stream>>>(Ws, wsq, n);
    n = NLAYERS * SC * RC;    k_prep_cvt<<<(n + 255) / 256, 256, 0, stream>>>(Wr, wrq, n);

    n = B * T; k_first<<<(n + 255) / 256, 256, 0, stream>>>(x, W_first, b_first, h0, skips, n);

    unsigned short* hb[2] = {h0, h1};
    for (int i = 0; i < NLAYERS; ++i) {
        int d = 1 << (i % 10);
        k_layer<<<dim3(B * T / TT), 128, 0, stream>>>(
            hb[i & 1], hb[(i + 1) & 1], skips, cup,
            wdq + (size_t)i * GC * 192,
            waq + (size_t)i * GC * ACPAD,
            wsq + (size_t)i * SC * RC,
            wrq + (size_t)i * SC * RC,
            bd + i * GC, bs + i * SC, br + i * SC, d, T);
    }

    n = B * T; k_final<<<(n + 255) / 256, 256, 0, stream>>>(skips, Wl1, bl1, Wl2, bl2, out, n);
}
